// SchNetInteraction_66666482368669
// MI455X (gfx1250) — matrix operand order repaired and checked by reading, not compiled, not run on hardware
//
#include <hip/hip_runtime.h>
#include <math.h>

#define DIM 128
#define TILE_M 16            // one WMMA tile of edges
#define EDGE_WAVES 4
#define NODE_WAVES 4
#define STRIDE (DIM + 8)     // bf16 elems; 272B row pitch keeps 16B align, spreads banks

typedef __attribute__((ext_vector_type(16))) __bf16    v16bf;
typedef __attribute__((ext_vector_type(8)))  __bf16    v8bf;
typedef __attribute__((ext_vector_type(8)))  float     v8f;
typedef __attribute__((ext_vector_type(8)))  unsigned  v8u;

// ---- bf16 helpers (round-half-up; one v_perm_b32 per pair) ----
__device__ __forceinline__ __bf16 f2bf(float f) {
  const unsigned u = __float_as_uint(f) + 0x8000u;
  const unsigned short s = (unsigned short)(u >> 16);
  return __builtin_bit_cast(__bf16, s);
}

__device__ __forceinline__ unsigned pack2bf(float x, float y) {
  const unsigned ux = __float_as_uint(x) + 0x8000u;
  const unsigned uy = __float_as_uint(y) + 0x8000u;
  return __builtin_amdgcn_perm(uy, ux, 0x07060302u);  // {y.hi16, x.hi16}
}

__device__ __forceinline__ v8f v8zero() {
  v8f z;
#pragma unroll
  for (int i = 0; i < 8; ++i) z[i] = 0.0f;
  return z;
}

// B-fragment (K=32 x N=16 bf16) from a row-major f32 row.
// B layout: lane = col n (lane&15); lanes 0-15 hold K=k0..k0+15,
// lanes 16-31 hold K=k0+16..k0+31 -> 16 CONSECUTIVE floats per lane.
__device__ __forceinline__ v16bf load_b_frag_f32(const float* row, int k0) {
  const float4 f0 = *(const float4*)(row + k0);
  const float4 f1 = *(const float4*)(row + k0 + 4);
  const float4 f2 = *(const float4*)(row + k0 + 8);
  const float4 f3 = *(const float4*)(row + k0 + 12);
  v8u p;
  p[0] = pack2bf(f0.x, f0.y); p[1] = pack2bf(f0.z, f0.w);
  p[2] = pack2bf(f1.x, f1.y); p[3] = pack2bf(f1.z, f1.w);
  p[4] = pack2bf(f2.x, f2.y); p[5] = pack2bf(f2.z, f2.w);
  p[6] = pack2bf(f3.x, f3.y); p[7] = pack2bf(f3.z, f3.w);
  return __builtin_bit_cast(v16bf, p);
}

// B-fragment from bf16 LDS staging: two aligned 16B loads, zero VALU.
__device__ __forceinline__ v16bf load_b_frag_lds(const __bf16* row, int k0) {
  const v8bf lo = *(const v8bf*)(row + k0);
  const v8bf hi = *(const v8bf*)(row + k0 + 8);
  return __builtin_shufflevector(lo, hi, 0, 1, 2, 3, 4, 5, 6, 7,
                                 8, 9, 10, 11, 12, 13, 14, 15);
}

// pack 8 f32 accumulator values -> 8 bf16, one b128 LDS store
__device__ __forceinline__ void store_frag8(__bf16* dst, const v8f a) {
  const uint4 q = make_uint4(pack2bf(a[0], a[1]), pack2bf(a[2], a[3]),
                             pack2bf(a[4], a[5]), pack2bf(a[6], a[7]));
  *(uint4*)dst = q;
}

// bias + SiLU on 8 accumulator values (v_rcp instead of IEEE divide)
__device__ __forceinline__ void bias_silu8(v8f& a, const float4 ba, const float4 bb) {
  a[0] += ba.x; a[1] += ba.y; a[2] += ba.z; a[3] += ba.w;
  a[4] += bb.x; a[5] += bb.y; a[6] += bb.z; a[7] += bb.w;
#pragma unroll
  for (int i = 0; i < 8; ++i) a[i] *= __builtin_amdgcn_rcpf(1.0f + __expf(-a[i]));
}

// gfx1250 async byte-copy global -> LDS (ASYNCcnt-tracked), 16B per lane per issue.
__device__ __forceinline__ void async_copy_to_lds(__bf16* lds_dst, const __bf16* gsrc,
                                                  int bytes, int tid, int nthreads) {
  const unsigned lbase = (unsigned)(size_t)(void*)lds_dst;  // low 32 bits = LDS offset
  for (int off = tid * 16; off < bytes; off += nthreads * 16) {
    const unsigned laddr = lbase + (unsigned)off;
    const unsigned goff  = (unsigned)off;
    asm volatile("global_load_async_to_lds_b128 %0, %1, %2 offset:0"
                 :: "v"(laddr), "v"(goff), "s"(gsrc) : "memory");
  }
}

__device__ __forceinline__ void wait_async_and_barrier() {
  asm volatile("s_wait_asynccnt 0" ::: "memory");
  __syncthreads();
}

// ---- one-time weight conversion into A-fragment layout (bf16) ----
// Transposed products (Dᵀ = W · actᵀ), so W is the A operand:
// A-frag(kc,mb): lane l holds row m = mb*16+(l&15); element e holds
// K = kc*32 + (l>>4)*16 + e; value = W[m][k].
__global__ void prep_kernel(const float* __restrict__ Wl, const float* __restrict__ W0,
                            const float* __restrict__ W2, __bf16* __restrict__ frag) {
  const int idx = blockIdx.x * blockDim.x + threadIdx.x;
  if (idx >= DIM * DIM) return;
  const int e  = idx & 15;
  const int l  = (idx >> 4) & 31;
  const int f  = idx >> 9;                 // kc*8 + mb
  const int kc = f >> 3;
  const int mb = f & 7;
  const int m  = mb * 16 + (l & 15);
  const int k  = kc * 32 + ((l >> 4) << 4) + e;
  frag[idx]                 = f2bf(Wl[m * DIM + k]);
  frag[idx + DIM * DIM]     = f2bf(W0[m * DIM + k]);
  frag[idx + 2 * DIM * DIM] = f2bf(W2[m * DIM + k]);
}

__global__ void zero_kernel(float* __restrict__ p, long long n) {
  const long long i = ((long long)blockIdx.x * blockDim.x + threadIdx.x) * 4;
  if (i + 3 < n) *(float4*)(p + i) = make_float4(0.f, 0.f, 0.f, 0.f);
}

// Wxhᵀ = Wl · xᵀ  (one wave -> 16 nodes x 128 feats; stored row-major [node][feat])
__global__ __launch_bounds__(NODE_WAVES * 32)
void node_kernel(const float* __restrict__ x, const __bf16* __restrict__ wfrag,
                 const float* __restrict__ bl, float* __restrict__ Wxh, int n_nodes) {
  __shared__ __align__(32) __bf16 Wlf[DIM * DIM];
  async_copy_to_lds(Wlf, wfrag, DIM * DIM * 2, threadIdx.x, blockDim.x);
  wait_async_and_barrier();

  const int wave = threadIdx.x >> 5;
  const int lane = threadIdx.x & 31;
  const int base = (blockIdx.x * NODE_WAVES + wave) * TILE_M;
  if (base >= n_nodes) return;

  const int ln15 = lane & 15;
  const int hi   = lane >> 4;

  int nrow = base + ln15;
  const bool active = nrow < n_nodes;
  if (!active) nrow = n_nodes - 1;
  const float* arow = x + (size_t)nrow * DIM;

  v8f acc[8];
#pragma unroll
  for (int mb = 0; mb < 8; ++mb) acc[mb] = v8zero();

#pragma unroll
  for (int kc = 0; kc < 4; ++kc) {
    const v16bf b = load_b_frag_f32(arow, kc * 32 + 16 * hi);
#pragma unroll
    for (int mb = 0; mb < 8; ++mb) {
      const v16bf a = *(const v16bf*)&Wlf[((kc * 8 + mb) * 32 + lane) * 16];
      acc[mb] = __builtin_amdgcn_wmma_f32_16x16x32_bf16(
          false, a, false, b, (short)0, acc[mb], false, false);
    }
  }
  // Dᵀ layout: lane = node col (lane&15); reg r -> feature d = mb*16 + 8*hi + r
  if (active) {
    float* orow = Wxh + (size_t)(base + ln15) * DIM;
#pragma unroll
    for (int mb = 0; mb < 8; ++mb) {
      const int d0 = mb * 16 + 8 * hi;
      const float4 ba = *(const float4*)(bl + d0);
      const float4 bb = *(const float4*)(bl + d0 + 4);
      *(float4*)(orow + d0)     = make_float4(acc[mb][0] + ba.x, acc[mb][1] + ba.y,
                                              acc[mb][2] + ba.z, acc[mb][3] + ba.w);
      *(float4*)(orow + d0 + 4) = make_float4(acc[mb][4] + bb.x, acc[mb][5] + bb.y,
                                              acc[mb][6] + bb.z, acc[mb][7] + bb.w);
    }
  }
}

// Fused edge pipeline (transposed), M=32 edges per wave (two 16-edge sub-tiles)
// so each weight A-fragment load from LDS feeds TWO WMMAs.
__global__ __launch_bounds__(EDGE_WAVES * 32)
void edge_kernel(const float* __restrict__ rbf,
                 const int* __restrict__ idx_i, const int* __restrict__ idx_j,
                 const __bf16* __restrict__ wfrag,  // W0 frags | W2 frags
                 const float* __restrict__ b0, const float* __restrict__ b2,
                 const float* __restrict__ Wxh, float* __restrict__ agg, int n_edges) {
  __shared__ __align__(32) __bf16 Wf[2 * DIM * DIM];              // 64 KB
  __shared__ __align__(16) __bf16 stag[EDGE_WAVES][2 * TILE_M * STRIDE];

  async_copy_to_lds(Wf, wfrag, 2 * DIM * DIM * 2, threadIdx.x, blockDim.x);
  wait_async_and_barrier();

  const int wave = threadIdx.x >> 5;
  const int lane = threadIdx.x & 31;
  const int base = (blockIdx.x * EDGE_WAVES + wave) * (2 * TILE_M);
  if (base >= n_edges) return;   // wave-uniform exit: WMMA runs with EXEC all-1s

  __bf16* st = stag[wave];       // wave-private staging; same-wave DS ops are in-order
  const int ln15 = lane & 15;
  const int hi   = lane >> 4;

  int e0 = base + ln15;                     // sub-tile 0 edge for this lane
  int e1 = base + TILE_M + ln15;            // sub-tile 1 edge for this lane
  const bool act0 = e0 < n_edges;
  const bool act1 = e1 < n_edges;
  if (!act0) e0 = n_edges - 1;
  if (!act1) e1 = n_edges - 1;
  const int j0 = idx_j[e0], i0 = idx_i[e0];
  const int j1 = idx_j[e1], i1 = idx_i[e1];
  const __bf16* srow0 = st + ln15 * STRIDE;
  const __bf16* srow1 = st + (TILE_M + ln15) * STRIDE;
  __bf16* swrow0 = st + ln15 * STRIDE;
  __bf16* swrow1 = st + (TILE_M + ln15) * STRIDE;

  v8f acc0[8], acc1[8];

  // ---------- stage 1: m1ᵀ = W0 · rbfᵀ ----------
#pragma unroll
  for (int mb = 0; mb < 8; ++mb) { acc0[mb] = v8zero(); acc1[mb] = v8zero(); }
  {
    const float* ar0 = rbf + (size_t)e0 * DIM;
    const float* ar1 = rbf + (size_t)e1 * DIM;
#pragma unroll
    for (int kc = 0; kc < 4; ++kc) {
      const v16bf bA = load_b_frag_f32(ar0, kc * 32 + 16 * hi);
      const v16bf bB = load_b_frag_f32(ar1, kc * 32 + 16 * hi);
#pragma unroll
      for (int mb = 0; mb < 8; ++mb) {
        const v16bf a = *(const v16bf*)&Wf[((kc * 8 + mb) * 32 + lane) * 16];
        acc0[mb] = __builtin_amdgcn_wmma_f32_16x16x32_bf16(
            false, a, false, bA, (short)0, acc0[mb], false, false);
        acc1[mb] = __builtin_amdgcn_wmma_f32_16x16x32_bf16(
            false, a, false, bB, (short)0, acc1[mb], false, false);
      }
    }
  }
  // boundary: lane holds 8 consecutive features per mb -> one b128 store each
#pragma unroll
  for (int mb = 0; mb < 8; ++mb) {
    const int d0 = mb * 16 + 8 * hi;
    store_frag8(swrow0 + d0, acc0[mb]);
    store_frag8(swrow1 + d0, acc1[mb]);
  }

  // ---------- stage 2: tᵀ = silu(W0 · m1ᵀ + b0) ----------
#pragma unroll
  for (int mb = 0; mb < 8; ++mb) { acc0[mb] = v8zero(); acc1[mb] = v8zero(); }
#pragma unroll
  for (int kc = 0; kc < 4; ++kc) {
    const v16bf bA = load_b_frag_lds(srow0, kc * 32 + 16 * hi);
    const v16bf bB = load_b_frag_lds(srow1, kc * 32 + 16 * hi);
#pragma unroll
    for (int mb = 0; mb < 8; ++mb) {
      const v16bf a = *(const v16bf*)&Wf[((kc * 8 + mb) * 32 + lane) * 16];
      acc0[mb] = __builtin_amdgcn_wmma_f32_16x16x32_bf16(
          false, a, false, bA, (short)0, acc0[mb], false, false);
      acc1[mb] = __builtin_amdgcn_wmma_f32_16x16x32_bf16(
          false, a, false, bB, (short)0, acc1[mb], false, false);
    }
  }
#pragma unroll
  for (int mb = 0; mb < 8; ++mb) {
    const int d0 = mb * 16 + 8 * hi;
    const float4 ba = *(const float4*)(b0 + d0);
    const float4 bb = *(const float4*)(b0 + d0 + 4);
    bias_silu8(acc0[mb], ba, bb);
    bias_silu8(acc1[mb], ba, bb);
    store_frag8(swrow0 + d0, acc0[mb]);
    store_frag8(swrow1 + d0, acc1[mb]);
  }

  // ---------- stage 3: mᵀ = W2 · tᵀ ----------
#pragma unroll
  for (int mb = 0; mb < 8; ++mb) { acc0[mb] = v8zero(); acc1[mb] = v8zero(); }
#pragma unroll
  for (int kc = 0; kc < 4; ++kc) {
    const v16bf bA = load_b_frag_lds(srow0, kc * 32 + 16 * hi);
    const v16bf bB = load_b_frag_lds(srow1, kc * 32 + 16 * hi);
#pragma unroll
    for (int mb = 0; mb < 8; ++mb) {
      const v16bf a = *(const v16bf*)&Wf[DIM * DIM + ((kc * 8 + mb) * 32 + lane) * 16];
      acc0[mb] = __builtin_amdgcn_wmma_f32_16x16x32_bf16(
          false, a, false, bA, (short)0, acc0[mb], false, false);
      acc1[mb] = __builtin_amdgcn_wmma_f32_16x16x32_bf16(
          false, a, false, bB, (short)0, acc1[mb], false, false);
    }
  }

  // ---------- finalize: (m + b2) * Wxh[j] -> atomic agg[i] ----------
  const float* w0r = Wxh + (size_t)j0 * DIM;
  const float* w1r = Wxh + (size_t)j1 * DIM;
  float* g0r = agg + (size_t)i0 * DIM;
  float* g1r = agg + (size_t)i1 * DIM;
#pragma unroll
  for (int mb = 0; mb < 8; ++mb) {
    const int d0 = mb * 16 + 8 * hi;
    const float4 ba = *(const float4*)(b2 + d0);
    const float4 bb = *(const float4*)(b2 + d0 + 4);
    if (act0) {
      const float4 wa = *(const float4*)(w0r + d0);
      const float4 wb = *(const float4*)(w0r + d0 + 4);
      atomicAdd(&g0r[d0 + 0], (acc0[mb][0] + ba.x) * wa.x);
      atomicAdd(&g0r[d0 + 1], (acc0[mb][1] + ba.y) * wa.y);
      atomicAdd(&g0r[d0 + 2], (acc0[mb][2] + ba.z) * wa.z);
      atomicAdd(&g0r[d0 + 3], (acc0[mb][3] + ba.w) * wa.w);
      atomicAdd(&g0r[d0 + 4], (acc0[mb][4] + bb.x) * wb.x);
      atomicAdd(&g0r[d0 + 5], (acc0[mb][5] + bb.y) * wb.y);
      atomicAdd(&g0r[d0 + 6], (acc0[mb][6] + bb.z) * wb.z);
      atomicAdd(&g0r[d0 + 7], (acc0[mb][7] + bb.w) * wb.w);
    }
    if (act1) {
      const float4 wa = *(const float4*)(w1r + d0);
      const float4 wb = *(const float4*)(w1r + d0 + 4);
      atomicAdd(&g1r[d0 + 0], (acc1[mb][0] + ba.x) * wa.x);
      atomicAdd(&g1r[d0 + 1], (acc1[mb][1] + ba.y) * wa.y);
      atomicAdd(&g1r[d0 + 2], (acc1[mb][2] + ba.z) * wa.z);
      atomicAdd(&g1r[d0 + 3], (acc1[mb][3] + ba.w) * wa.w);
      atomicAdd(&g1r[d0 + 4], (acc1[mb][4] + bb.x) * wb.x);
      atomicAdd(&g1r[d0 + 5], (acc1[mb][5] + bb.y) * wb.y);
      atomicAdd(&g1r[d0 + 6], (acc1[mb][6] + bb.z) * wb.z);
      atomicAdd(&g1r[d0 + 7], (acc1[mb][7] + bb.w) * wb.w);
    }
  }
}

// out = LayerNorm(x + agg) * gamma + beta   (one wave per row)
__global__ __launch_bounds__(256)
void ln_kernel(const float* __restrict__ x, const float* __restrict__ agg,
               const float* __restrict__ gamma, const float* __restrict__ beta,
               float* __restrict__ out, int n_nodes) {
  const int wave = threadIdx.x >> 5;
  const int lane = threadIdx.x & 31;
  const int row  = blockIdx.x * 8 + wave;
  if (row >= n_nodes) return;

  const size_t off = (size_t)row * DIM + lane * 4;
  const float4 xv = *(const float4*)(x + off);
  const float4 av = *(const float4*)(agg + off);
  const float h0 = xv.x + av.x, h1 = xv.y + av.y, h2 = xv.z + av.z, h3 = xv.w + av.w;

  float s  = h0 + h1 + h2 + h3;
  float sq = h0 * h0 + h1 * h1 + h2 * h2 + h3 * h3;
#pragma unroll
  for (int o = 16; o > 0; o >>= 1) {
    s  += __shfl_xor(s, o, 32);
    sq += __shfl_xor(sq, o, 32);
  }
  const float mu   = s * (1.0f / DIM);
  const float var  = sq * (1.0f / DIM) - mu * mu;
  const float rinv = rsqrtf(var + 1e-5f);

  const float4 g = *(const float4*)(gamma + lane * 4);
  const float4 b = *(const float4*)(beta + lane * 4);
  float4 o4;
  o4.x = (h0 - mu) * rinv * g.x + b.x;
  o4.y = (h1 - mu) * rinv * g.y + b.y;
  o4.z = (h2 - mu) * rinv * g.z + b.z;
  o4.w = (h3 - mu) * rinv * g.w + b.w;
  *(float4*)(out + off) = o4;
}

extern "C" void kernel_launch(void* const* d_in, const int* in_sizes, int n_in,
                              void* d_out, int out_size, void* d_ws, size_t ws_size,
                              hipStream_t stream) {
  const float* x     = (const float*)d_in[0];
  const int*   ii    = (const int*)d_in[1];
  const int*   jj    = (const int*)d_in[2];
  const float* rbf   = (const float*)d_in[3];
  const float* Wl    = (const float*)d_in[4];
  const float* bl    = (const float*)d_in[5];
  const float* W0    = (const float*)d_in[6];
  const float* b0    = (const float*)d_in[7];
  const float* W2    = (const float*)d_in[8];
  const float* b2    = (const float*)d_in[9];
  const float* gamma = (const float*)d_in[10];
  const float* beta  = (const float*)d_in[11];

  const int n_nodes = in_sizes[0] / DIM;
  const int n_edges = in_sizes[1];

  // workspace layout: [bf16 weight A-fragments: Wl | W0 | W2][Wxh f32][agg f32]
  __bf16* frag = (__bf16*)d_ws;                              // 3 * 16384 bf16 = 96 KB
  float*  Wxh  = (float*)((char*)d_ws + 3 * DIM * DIM * sizeof(__bf16));
  float*  agg  = Wxh + (size_t)n_nodes * DIM;
  float*  out  = (float*)d_out;

  // one-time weight conversion into A-fragment layout
  prep_kernel<<<(DIM * DIM + 255) / 256, 256, 0, stream>>>(Wl, W0, W2, frag);

  // zero the scatter accumulator (harness does not re-zero between replays)
  const long long agg_elems = (long long)n_nodes * DIM;
  const int zt = 256;
  const int zb = (int)((agg_elems / 4 + zt - 1) / zt);
  zero_kernel<<<zb, zt, 0, stream>>>(agg, agg_elems);

  // Wxh = x @ Wl^T + bl
  const int node_tiles = (n_nodes + TILE_M - 1) / TILE_M;
  const int node_wgs   = (node_tiles + NODE_WAVES - 1) / NODE_WAVES;
  node_kernel<<<node_wgs, NODE_WAVES * 32, 0, stream>>>(x, frag, bl, Wxh, n_nodes);

  // fused edge pipeline + scatter (32 edges per wave)
  const int edge_tiles = (n_edges + 2 * TILE_M - 1) / (2 * TILE_M);
  const int edge_wgs   = (edge_tiles + EDGE_WAVES - 1) / EDGE_WAVES;
  edge_kernel<<<edge_wgs, EDGE_WAVES * 32, 0, stream>>>(
      rbf, ii, jj, frag + DIM * DIM, b0, b2, Wxh, agg, n_edges);

  // residual + LayerNorm
  const int ln_blocks = (n_nodes + 7) / 8;
  ln_kernel<<<ln_blocks, 256, 0, stream>>>(x, agg, gamma, beta, out, n_nodes);
}
